// SlotAttention_1365799600885
// MI455X (gfx1250) — matrix operand order repaired and checked by reading, not compiled, not run on hardware
//
#include <hip/hip_runtime.h>

// ---------------------------------------------------------------------------
// Slot Attention on gfx1250 (MI455X).  B=64, N=4096, D=128, S=7, H=256, 3 iters.
// HBM-bound: K/V stored bf16, WMMA f32_16x16x32_bf16 for all large GEMMs.
// ---------------------------------------------------------------------------

#define B_  64
#define N_  4096
#define D_  128
#define S_  7
#define H_  256
#define EPS_ 1e-8f

typedef __attribute__((ext_vector_type(16))) __bf16 v16bf;
typedef __attribute__((ext_vector_type(8)))  float  v8f;

union Frag {
    v16bf v;
    unsigned short u[16];
    uint4 q[2];
};

__device__ __forceinline__ unsigned short f2bf(float f) {
    unsigned int u = __float_as_uint(f);
    u += 0x7FFFu + ((u >> 16) & 1u);           // round-to-nearest-even
    return (unsigned short)(u >> 16);
}
__device__ __forceinline__ float bf2f(unsigned short h) {
    return __uint_as_float(((unsigned int)h) << 16);
}

__device__ __forceinline__ v8f wmma_bf16(const Frag& a, const Frag& b, v8f c) {
    return __builtin_amdgcn_wmma_f32_16x16x32_bf16(
        /*neg_a=*/false, a.v, /*neg_b=*/false, b.v,
        /*c_mod=*/(short)0, c, /*reuse_a=*/false, /*reuse_b=*/false);
}

// ---------------------------------------------------------------------------
// Kernel 1: fused LayerNorm(embeddings) -> K = ln@Wk^T+bk, V^T (transposed).
// 64 rows per block, 4 waves (16 rows each).  K: [B*N, D] bf16 row-major.
// Vt: [B, D, N] bf16 (so updates-GEMM B-fragments are contiguous per lane).
// ---------------------------------------------------------------------------
__global__ __launch_bounds__(128) void ln_kv_kernel(
    const float* __restrict__ emb,
    const float* __restrict__ Wk, const float* __restrict__ bk,
    const float* __restrict__ Wv, const float* __restrict__ bv,
    const float* __restrict__ g,  const float* __restrict__ be,
    unsigned short* __restrict__ K, unsigned short* __restrict__ Vt)
{
    extern __shared__ char smem[];
    float*          s_emb = (float*)smem;                         // 64*128 f32  (32768 B)
    unsigned short* s_a   = (unsigned short*)(smem + 32768);      // 64*128 bf16 (16384 B)
    unsigned short* s_wk  = (unsigned short*)(smem + 49152);      // 128*128 bf16 (32768 B)
    unsigned short* s_wv  = (unsigned short*)(smem + 81920);      // 128*128 bf16 (32768 B)
    float*          s_bk  = (float*)(smem + 114688);
    float*          s_bv  = (float*)(smem + 115200);
    float*          s_g   = (float*)(smem + 115712);
    float*          s_be  = (float*)(smem + 116224);

    const int tid = threadIdx.x;
    const long long row0 = (long long)blockIdx.x * 64;
    const int bb    = (int)(row0 >> 12);      // batch (64 | 4096)
    const int tbase = (int)(row0 & (N_ - 1)); // token base within batch

    // Stage embeddings tile (64 x 128 f32) with 128-bit loads.
    {
        const float4* src = (const float4*)(emb + row0 * D_);
        float4* dst = (float4*)s_emb;
        for (int i = tid; i < 64 * 32; i += 128) dst[i] = src[i];
    }
    // Stage Wk / Wv as bf16.
    for (int i = tid; i < 4096; i += 128) {
        float4 a = ((const float4*)Wk)[i];
        float4 b = ((const float4*)Wv)[i];
        ushort4 ka, vb;
        ka.x = f2bf(a.x); ka.y = f2bf(a.y); ka.z = f2bf(a.z); ka.w = f2bf(a.w);
        vb.x = f2bf(b.x); vb.y = f2bf(b.y); vb.z = f2bf(b.z); vb.w = f2bf(b.w);
        ((ushort4*)s_wk)[i] = ka;
        ((ushort4*)s_wv)[i] = vb;
    }
    if (tid < 128) { s_bk[tid] = bk[tid]; s_bv[tid] = bv[tid]; s_g[tid] = g[tid]; s_be[tid] = be[tid]; }
    __syncthreads();

    const int wave = tid >> 5, lane = tid & 31, rl = lane & 15, half = lane >> 4;
    const int lrow = wave * 16 + rl;

    // LayerNorm: 2 lanes per row, each sums 64 elems, combine via shfl_xor(16).
    {
        const float* xr = s_emb + lrow * D_ + half * 64;
        float sum = 0.f, sq = 0.f;
        for (int c = 0; c < 64; ++c) { float x = xr[c]; sum += x; sq += x * x; }
        sum += __shfl_xor(sum, 16, 32);
        sq  += __shfl_xor(sq, 16, 32);
        float mean = sum * (1.f / 128.f);
        float var  = sq * (1.f / 128.f) - mean * mean;
        float rstd = rsqrtf(var + 1e-5f);
        unsigned short* ar = s_a + lrow * D_;
        for (int c = 0; c < 64; ++c) {
            int d = half * 64 + c;
            ar[d] = f2bf((xr[c] - mean) * rstd * s_g[d] + s_be[d]);
        }
    }
    __syncthreads();

    // A fragments (16x32 bf16, contiguous k order: lane half h holds k = 16h..16h+15, as B).
    Frag A[4];
    {
        const unsigned short* ar = s_a + (wave * 16 + rl) * D_;
        for (int kc = 0; kc < 4; ++kc) {
            A[kc].q[0] = *(const uint4*)(ar + kc * 32 + 16 * half);
            A[kc].q[1] = *(const uint4*)(ar + kc * 32 + 16 * half + 8);
        }
    }

    for (int nt = 0; nt < 8; ++nt) {
        const int col = nt * 16 + rl;
        const unsigned short* wkr = s_wk + col * D_;
        const unsigned short* wvr = s_wv + col * D_;
        v8f accK, accV;
        const float bkv = s_bk[col], bvv = s_bv[col];
        for (int i = 0; i < 8; ++i) { accK[i] = bkv; accV[i] = bvv; }
        for (int kc = 0; kc < 4; ++kc) {
            Frag Bk, Bv;
            Bk.q[0] = *(const uint4*)(wkr + kc * 32 + 16 * half);
            Bk.q[1] = *(const uint4*)(wkr + kc * 32 + 16 * half + 8);
            Bv.q[0] = *(const uint4*)(wvr + kc * 32 + 16 * half);
            Bv.q[1] = *(const uint4*)(wvr + kc * 32 + 16 * half + 8);
            accK = wmma_bf16(A[kc], Bk, accK);
            accV = wmma_bf16(A[kc], Bv, accV);
        }
        // K row-major store: C/D layout -> row m = v + 8*half, col = nt*16 + rl.
        {
            const long long gr0 = row0 + wave * 16 + 8 * half;
            for (int v = 0; v < 8; ++v)
                K[(gr0 + v) * D_ + col] = f2bf(accK[v]);
        }
        // V transposed store: Vt[b][d][t], per-lane 8 consecutive t -> pack pairs.
        {
            unsigned short* vrow = Vt + ((size_t)bb * D_ + col) * N_
                                      + tbase + wave * 16 + 8 * half;
            for (int v = 0; v < 8; v += 2) {
                unsigned int pk = (unsigned int)f2bf(accV[v])
                                | ((unsigned int)f2bf(accV[v + 1]) << 16);
                *(unsigned int*)(vrow + v) = pk;
            }
        }
    }
}

// ---------------------------------------------------------------------------
// Kernel 2: q = LN(slots) @ Wq^T + bq  (tiny: 448 valid rows).  q stored bf16
// padded to 16 rows per batch (rows s>=7 zeroed).
// ---------------------------------------------------------------------------
__global__ __launch_bounds__(128) void qproj_kernel(
    const float* __restrict__ slots,
    const float* __restrict__ Wq, const float* __restrict__ bq,
    const float* __restrict__ g,  const float* __restrict__ be,
    unsigned short* __restrict__ q)
{
    __shared__ float s_x[128];
    __shared__ float s_part[4];
    const int bs = blockIdx.x, b = bs >> 4, s = bs & 15, j = threadIdx.x;
    if (s >= S_) { q[(size_t)bs * D_ + j] = 0; return; }
    float x = slots[((size_t)b * S_ + s) * D_ + j];

    float sum = x;
    for (int m = 16; m >= 1; m >>= 1) sum += __shfl_xor(sum, m, 32);
    if ((j & 31) == 0) s_part[j >> 5] = sum;
    __syncthreads();
    float mean = (s_part[0] + s_part[1] + s_part[2] + s_part[3]) * (1.f / 128.f);
    __syncthreads();
    float dx = x - mean, sq = dx * dx;
    for (int m = 16; m >= 1; m >>= 1) sq += __shfl_xor(sq, m, 32);
    if ((j & 31) == 0) s_part[j >> 5] = sq;
    __syncthreads();
    float var  = (s_part[0] + s_part[1] + s_part[2] + s_part[3]) * (1.f / 128.f);
    float rstd = rsqrtf(var + 1e-5f);
    s_x[j] = dx * rstd * g[j] + be[j];
    __syncthreads();

    float acc = bq[j];
    const float* wr = Wq + (size_t)j * D_;
    for (int k = 0; k < D_; ++k) acc += s_x[k] * wr[k];
    q[(size_t)bs * D_ + j] = f2bf(acc);
}

// ---------------------------------------------------------------------------
// Kernel 3: per-batch attention.  One block = one batch, 16 waves (512 thr).
// Phase 1: dots = q @ K^T -> LDS (bf16, 16x4096 = 128 KB), 16 n-tiles/wave.
// Phase 2: softmax + analytic EPS-renorm (one row per wave).
// Phase 3: updates = attn @ V.  K-reduction split across wave pairs, two
//          interleaved accumulators per wave (dependent chain 128 -> 32),
//          partials combined through LDS.
// ---------------------------------------------------------------------------
__global__ __launch_bounds__(512) void attn_kernel(
    const unsigned short* __restrict__ q,
    const unsigned short* __restrict__ K,
    const unsigned short* __restrict__ Vt,
    float* __restrict__ updates)
{
    extern __shared__ char smem[];
    unsigned short* s_attn = (unsigned short*)smem;   // 16 x 4096 bf16 (131072 B)
    float* s_part = (float*)(smem + 131072);          // 8 x 16 x 16 f32 (8192 B)

    const int b = blockIdx.x;
    const int tid = threadIdx.x, wave = tid >> 5, lane = tid & 31;
    const int rl = lane & 15, half = lane >> 4;

    // ---- Phase 1: dots -------------------------------------------------
    Frag A[4];
    {
        const unsigned short* qr = q + ((size_t)b * 16 + rl) * D_;
        for (int kc = 0; kc < 4; ++kc) {
            A[kc].q[0] = *(const uint4*)(qr + kc * 32 + 16 * half);
            A[kc].q[1] = *(const uint4*)(qr + kc * 32 + 16 * half + 8);
        }
    }
    const unsigned short* kb = K + (size_t)b * N_ * D_;
    for (int j = 0; j < 16; ++j) {                    // 16 n-tiles per wave
        const int n0 = (wave * 16 + j) * 16;
        const unsigned short* kr = kb + (size_t)(n0 + rl) * D_;
        __builtin_prefetch(kr + 16 * D_, 0, 0);       // next n-tile row (global_prefetch_b8)
        v8f acc;
        for (int i = 0; i < 8; ++i) acc[i] = 0.f;
        for (int kc = 0; kc < 4; ++kc) {
            Frag Bf;
            Bf.q[0] = *(const uint4*)(kr + kc * 32 + 16 * half);
            Bf.q[1] = *(const uint4*)(kr + kc * 32 + 16 * half + 8);
            acc = wmma_bf16(A[kc], Bf, acc);
        }
        for (int v = 0; v < 8; ++v)
            s_attn[(v + 8 * half) * N_ + n0 + rl] = f2bf(acc[v]);
    }
    __syncthreads();

    // ---- Phase 2: softmax rows (one row per wave) ----------------------
    {
        unsigned short* rowp = s_attn + wave * N_;
        float mx = -3.0e38f;
        for (int i = lane; i < N_; i += 32) mx = fmaxf(mx, bf2f(rowp[i]));
        for (int m = 16; m >= 1; m >>= 1) mx = fmaxf(mx, __shfl_xor(mx, m, 32));
        float sum = 0.f;
        for (int i = lane; i < N_; i += 32) {
            float p = __expf(bf2f(rowp[i]) - mx);
            rowp[i] = f2bf(p);
            sum += p;
        }
        for (int m = 16; m >= 1; m >>= 1) sum += __shfl_xor(sum, m, 32);
        const float scale = 1.f / (sum + (float)N_ * EPS_);  // softmax row sums to `sum`
        for (int i = lane; i < N_; i += 32)
            rowp[i] = f2bf((bf2f(rowp[i]) + EPS_) * scale);
    }
    __syncthreads();

    // ---- Phase 3: updates ----------------------------------------------
    {
        const int dtile = wave & 7;
        const int ksel  = wave >> 3;                  // which half of K-range
        const int d = dtile * 16 + rl;
        const unsigned short* vr = Vt + ((size_t)b * D_ + d) * N_;
        const unsigned short* ar = s_attn + rl * N_;
        const int kc0 = ksel * 64;
        v8f acc0, acc1;
        for (int i = 0; i < 8; ++i) { acc0[i] = 0.f; acc1[i] = 0.f; }
        for (int kc = kc0; kc < kc0 + 64; kc += 2) {
            __builtin_prefetch(vr + (kc + 8) * 32, 0, 0);
            Frag Af0, Bf0, Af1, Bf1;
            Af0.q[0] = *(const uint4*)(ar + kc * 32 + 16 * half);
            Af0.q[1] = *(const uint4*)(ar + kc * 32 + 16 * half + 8);
            Bf0.q[0] = *(const uint4*)(vr + kc * 32 + 16 * half);
            Bf0.q[1] = *(const uint4*)(vr + kc * 32 + 16 * half + 8);
            Af1.q[0] = *(const uint4*)(ar + (kc + 1) * 32 + 16 * half);
            Af1.q[1] = *(const uint4*)(ar + (kc + 1) * 32 + 16 * half + 8);
            Bf1.q[0] = *(const uint4*)(vr + (kc + 1) * 32 + 16 * half);
            Bf1.q[1] = *(const uint4*)(vr + (kc + 1) * 32 + 16 * half + 8);
            acc0 = wmma_bf16(Af0, Bf0, acc0);
            acc1 = wmma_bf16(Af1, Bf1, acc1);
        }
        for (int i = 0; i < 8; ++i) acc0[i] += acc1[i];

        if (ksel == 1) {
            for (int v = 0; v < 8; ++v)
                s_part[dtile * 256 + (v + 8 * half) * 16 + rl] = acc0[v];
        }
        __syncthreads();
        if (ksel == 0) {
            for (int v = 0; v < 8; ++v) {
                float r = acc0[v] + s_part[dtile * 256 + (v + 8 * half) * 16 + rl];
                updates[((size_t)b * 16 + v + 8 * half) * D_ + d] = r;
            }
        }
    }
}

// ---------------------------------------------------------------------------
// Kernel 4: GRU cell + LN + MLP residual.  One block per valid (b,s) row.
// 0.15 GFLOP total -> plain VALU.
// ---------------------------------------------------------------------------
__device__ __forceinline__ float dot128(const float* __restrict__ w,
                                        const float* __restrict__ x) {
    float a = 0.f;
    for (int k = 0; k < 128; ++k) a += w[k] * x[k];
    return a;
}

__global__ __launch_bounds__(128) void gruff_kernel(
    const float* __restrict__ updates,
    const float* __restrict__ W_ih, const float* __restrict__ W_hh,
    const float* __restrict__ b_ih, const float* __restrict__ b_hh,
    const float* __restrict__ W1,  const float* __restrict__ b1,
    const float* __restrict__ W2,  const float* __restrict__ b2,
    const float* __restrict__ g,   const float* __restrict__ be,
    float* __restrict__ slots)
{
    __shared__ float s_x[128], s_h[128], s_a1[256], s_part[4];
    const int bs = blockIdx.x, b = bs / S_, s = bs % S_, j = threadIdx.x;

    float x = updates[((size_t)b * 16 + s) * D_ + j];
    float h = slots[((size_t)b * S_ + s) * D_ + j];
    s_x[j] = x; s_h[j] = h;
    __syncthreads();

    float gir = b_ih[j]        + dot128(W_ih + (size_t)j * D_, s_x);
    float giz = b_ih[128 + j]  + dot128(W_ih + (size_t)(128 + j) * D_, s_x);
    float gin = b_ih[256 + j]  + dot128(W_ih + (size_t)(256 + j) * D_, s_x);
    float ghr = b_hh[j]        + dot128(W_hh + (size_t)j * D_, s_h);
    float ghz = b_hh[128 + j]  + dot128(W_hh + (size_t)(128 + j) * D_, s_h);
    float ghn = b_hh[256 + j]  + dot128(W_hh + (size_t)(256 + j) * D_, s_h);

    float r = 1.f / (1.f + __expf(-(gir + ghr)));
    float z = 1.f / (1.f + __expf(-(giz + ghz)));
    float n = tanhf(gin + r * ghn);
    float snew = (1.f - z) * n + z * h;
    __syncthreads();

    // LayerNorm(snew)
    float sum = snew;
    for (int m = 16; m >= 1; m >>= 1) sum += __shfl_xor(sum, m, 32);
    if ((j & 31) == 0) s_part[j >> 5] = sum;
    __syncthreads();
    float mean = (s_part[0] + s_part[1] + s_part[2] + s_part[3]) * (1.f / 128.f);
    __syncthreads();
    float dx = snew - mean, sq = dx * dx;
    for (int m = 16; m >= 1; m >>= 1) sq += __shfl_xor(sq, m, 32);
    if ((j & 31) == 0) s_part[j >> 5] = sq;
    __syncthreads();
    float var  = (s_part[0] + s_part[1] + s_part[2] + s_part[3]) * (1.f / 128.f);
    float rstd = rsqrtf(var + 1e-5f);
    s_x[j] = dx * rstd * g[j] + be[j];
    __syncthreads();

    for (int t = j; t < H_; t += 128) {
        float a = b1[t] + dot128(W1 + (size_t)t * D_, s_x);
        s_a1[t] = fmaxf(a, 0.f);
    }
    __syncthreads();

    float o = b2[j];
    const float* w2r = W2 + (size_t)j * H_;
    for (int k = 0; k < H_; ++k) o += s_a1[k] * w2r[k];
    slots[((size_t)b * S_ + s) * D_ + j] = snew + o;
}

// ---------------------------------------------------------------------------
// Tiny init / copy kernels.
// ---------------------------------------------------------------------------
__global__ void init_kernel(const float* __restrict__ noise,
                            const float* __restrict__ mu,
                            const float* __restrict__ sigma,
                            float* __restrict__ slots) {
    int i = blockIdx.x * blockDim.x + threadIdx.x;
    if (i < B_ * S_ * D_) {
        int d = i & (D_ - 1);
        slots[i] = mu[d] + sigma[d] * noise[i];
    }
}

__global__ void copy_kernel(const float* __restrict__ slots, float* __restrict__ out) {
    int i = blockIdx.x * blockDim.x + threadIdx.x;
    if (i < B_ * S_ * D_) out[i] = slots[i];
}

// ---------------------------------------------------------------------------
extern "C" void kernel_launch(void* const* d_in, const int* in_sizes, int n_in,
                              void* d_out, int out_size, void* d_ws, size_t ws_size,
                              hipStream_t stream) {
    const float* emb     = (const float*)d_in[0];
    const float* noise   = (const float*)d_in[1];
    const float* mu      = (const float*)d_in[2];
    const float* sig     = (const float*)d_in[3];
    const float* Wk      = (const float*)d_in[4];
    const float* bk      = (const float*)d_in[5];
    const float* Wq      = (const float*)d_in[6];
    const float* bq      = (const float*)d_in[7];
    const float* Wv      = (const float*)d_in[8];
    const float* bv      = (const float*)d_in[9];
    const float* W_ih    = (const float*)d_in[10];
    const float* W_hh    = (const float*)d_in[11];
    const float* b_ih    = (const float*)d_in[12];
    const float* b_hh    = (const float*)d_in[13];
    const float* W1      = (const float*)d_in[14];
    const float* b1      = (const float*)d_in[15];
    const float* W2      = (const float*)d_in[16];
    const float* b2      = (const float*)d_in[17];
    const float* ln_in_g = (const float*)d_in[18];
    const float* ln_in_b = (const float*)d_in[19];
    const float* ln_s_g  = (const float*)d_in[20];
    const float* ln_s_b  = (const float*)d_in[21];
    const float* ln_ff_g = (const float*)d_in[22];
    const float* ln_ff_b = (const float*)d_in[23];

    // Workspace layout (~135 MB):
    unsigned short* K  = (unsigned short*)d_ws;                  // 64 MiB
    unsigned short* Vt = K + (size_t)B_ * N_ * D_;               // 64 MiB
    unsigned short* q  = Vt + (size_t)B_ * N_ * D_;              // 256 KiB (padded S=16)
    float* updates = (float*)(q + (size_t)B_ * 16 * D_);         // 512 KiB (padded S=16)
    float* slots   = updates + (size_t)B_ * 16 * D_;             // 224 KiB

    const int ntot = B_ * S_ * D_;
    init_kernel<<<(ntot + 255) / 256, 256, 0, stream>>>(noise, mu, sig, slots);

    ln_kv_kernel<<<(B_ * N_) / 64, 128, 116736, stream>>>(
        emb, Wk, bk, Wv, bv, ln_in_g, ln_in_b, K, Vt);

    for (int it = 0; it < 3; ++it) {
        qproj_kernel<<<B_ * 16, 128, 0, stream>>>(slots, Wq, bq, ln_s_g, ln_s_b, q);
        attn_kernel<<<B_, 512, 131072 + 8192, stream>>>(q, K, Vt, updates);
        gruff_kernel<<<B_ * S_, 128, 0, stream>>>(
            updates, W_ih, W_hh, b_ih, b_hh, W1, b1, W2, b2, ln_ff_g, ln_ff_b, slots);
    }

    copy_kernel<<<(ntot + 255) / 256, 256, 0, stream>>>(slots, (float*)d_out);
}
